// GAT_geom_11269994184787
// MI455X (gfx1250) — matrix operand order repaired and checked by reading, not compiled, not run on hardware
//
#include <hip/hip_runtime.h>

// ---- problem constants (match reference) ----
#define NNODES 20000
#define NEDGES 320000
#define ETOT   (NEDGES + NNODES)   // 340000 (edges + self loops)
#define DFEAT  32
#define NH1    8
#define CH     32                  // per-head channels (both layers)
#define F1     (NH1 * CH)          // 256
#define NEG_SLOPE 0.2f

typedef __attribute__((ext_vector_type(16))) _Float16 v16h;
typedef __attribute__((ext_vector_type(8)))  float    v8f;

// ---------------- utility ----------------
__global__ void fill_kernel(float* __restrict__ p, float v, int n) {
    int i = blockIdx.x * blockDim.x + threadIdx.x;
    if (i < n) p[i] = v;
}

__device__ inline void atomicMaxFloat(float* addr, float val) {
    // ordered-int trick; buffer pre-filled with -inf
    if (val >= 0.0f) atomicMax((int*)addr, __float_as_int(val));
    else             atomicMin((unsigned int*)addr, (unsigned int)__float_as_int(val));
}

__device__ inline void edge_nodes(int e, const int* __restrict__ src,
                                  const int* __restrict__ dst, int& s, int& d) {
    if (e < NEDGES) { s = src[e]; d = dst[e]; }
    else            { s = e - NEDGES; d = s; }   // appended self loops
}

// ---------------- GEMM 1: xh1[20000,256] = f16(X[20000,32]) @ f16(W1[32,256]) ----------------
// One 16x16 tile per wave; K=32 -> exactly one v_wmma_f32_16x16x32_f16.
__global__ void gemm1_wmma(const float* __restrict__ X,
                           const float* __restrict__ W,
                           float* __restrict__ Y) {
    int lane = threadIdx.x & 31;
    int wave = threadIdx.x >> 5;
    int tile = blockIdx.x * 8 + wave;       // 20000 tiles total, grid sized exactly
    int tm = tile >> 4;                     // 1250 row tiles
    int tn = tile & 15;                     // 16 col tiles
    int row0 = tm * 16, col0 = tn * 16;
    int grp = lane >> 4;

    v16h a, b;
    const float* xr = X + (row0 + (lane & 15)) * DFEAT;
#pragma unroll
    for (int h = 0; h < 16; ++h) {
        int k = grp * 8 + (h & 7) + ((h & 8) ? 16 : 0);   // 16-bit A 16x32 layout
        a[h] = (_Float16)xr[k];
    }
    int cB = col0 + (lane & 15);
#pragma unroll
    for (int h = 0; h < 16; ++h) {
        int k = grp * 8 + (h & 7) + ((h & 8) ? 16 : 0);   // 16-bit B 32x16 layout (table k order, matches A)
        b[h] = (_Float16)W[k * F1 + cB];
    }
    v8f c = {};
    c = __builtin_amdgcn_wmma_f32_16x16x32_f16(false, a, false, b, (short)0, c, false, false);
#pragma unroll
    for (int v = 0; v < 8; ++v) {
        Y[(row0 + v + grp * 8) * F1 + cB] = c[v];         // f32 C/D layout
    }
}

// ---------------- GEMM 2: xh2[20000,32] = X1h[20000,256](f16) @ f16(W2[256,32]) ----------------
__global__ void gemm2_wmma(const _Float16* __restrict__ X,
                           const float* __restrict__ W,
                           float* __restrict__ Y) {
    int lane = threadIdx.x & 31;
    int wave = threadIdx.x >> 5;
    int tile = blockIdx.x * 8 + wave;
    if (tile >= 1250 * 2) return;           // wave-uniform guard (EXEC stays all-1s)
    int tm = tile >> 1;
    int tn = tile & 1;
    int row0 = tm * 16, col0 = tn * 16;
    int grp = lane >> 4;
    int cB = col0 + (lane & 15);
    const _Float16* xr = X + (row0 + (lane & 15)) * F1;

    v8f c = {};
#pragma unroll
    for (int kk = 0; kk < 8; ++kk) {
        v16h a, b;
#pragma unroll
        for (int h = 0; h < 16; ++h) {
            int k = kk * 32 + grp * 8 + (h & 7) + ((h & 8) ? 16 : 0);
            a[h] = xr[k];
        }
#pragma unroll
        for (int h = 0; h < 16; ++h) {
            int k = kk * 32 + grp * 8 + (h & 7) + ((h & 8) ? 16 : 0);
            b[h] = (_Float16)W[k * CH + cB];
        }
        c = __builtin_amdgcn_wmma_f32_16x16x32_f16(false, a, false, b, (short)0, c, false, false);
    }
#pragma unroll
    for (int v = 0; v < 8; ++v) {
        Y[(row0 + v + grp * 8) * CH + cB] = c[v];
    }
}

// ---------------- attention scalars: alpha_src/dst[n,h] = sum_c xh[n,h,c]*att[h,c] ----------------
template <int HEADS>
__global__ void alpha_kernel(const float* __restrict__ XH,
                             const float* __restrict__ att_s,
                             const float* __restrict__ att_d,
                             float* __restrict__ as_out, float* __restrict__ ad_out) {
    int i = blockIdx.x * blockDim.x + threadIdx.x;
    if (i >= NNODES * HEADS) return;
    int node = i / HEADS, h = i % HEADS;
    const float* x = XH + (node * HEADS + h) * CH;
    float s = 0.f, d = 0.f;
#pragma unroll
    for (int c = 0; c < CH; ++c) {
        float xv = x[c];
        s += xv * att_s[h * CH + c];
        d += xv * att_d[h * CH + c];
    }
    as_out[i] = s;
    ad_out[i] = d;
}

// ---------------- edge pass 1: e = leakyrelu(as[src]+ad[dst]); segment max into m[dst] ----------------
template <int HEADS>
__global__ void edge_max_kernel(const int* __restrict__ src, const int* __restrict__ dst,
                                const float* __restrict__ as_in, const float* __restrict__ ad_in,
                                float* __restrict__ e_buf, float* __restrict__ m) {
    int i = blockIdx.x * blockDim.x + threadIdx.x;
    if (i >= ETOT * HEADS) return;
    int e = i / HEADS, h = i % HEADS;
    int s, d; edge_nodes(e, src, dst, s, d);
    float v = as_in[s * HEADS + h] + ad_in[d * HEADS + h];
    v = (v > 0.f) ? v : NEG_SLOPE * v;
    e_buf[i] = v;
    atomicMaxFloat(&m[d * HEADS + h], v);
}

// ---------------- edge pass 2: ex = exp(e - m[dst]); segment sum into den[dst] ----------------
template <int HEADS>
__global__ void edge_exp_kernel(const int* __restrict__ src, const int* __restrict__ dst,
                                float* __restrict__ e_buf,
                                const float* __restrict__ m, float* __restrict__ den) {
    int i = blockIdx.x * blockDim.x + threadIdx.x;
    if (i >= ETOT * HEADS) return;
    int e = i / HEADS, h = i % HEADS;
    int s, d; edge_nodes(e, src, dst, s, d);
    float ex = __expf(e_buf[i] - m[d * HEADS + h]);
    e_buf[i] = ex;
    atomicAdd(&den[d * HEADS + h], ex);
}

// ---------------- edge pass 3 (coalesced): one thread per (edge, feature) ----------------
// Wave32 covers 32 consecutive channels of a single edge: the xh[src] gather is one
// contiguous 128B line, the agg[dst] atomics hit one contiguous 128B region (L2-mergeable),
// and e_buf/den/src/dst reads are wave-uniform broadcasts.
template <int HEADS>
__global__ void edge_agg_kernel(const int* __restrict__ src, const int* __restrict__ dst,
                                const float* __restrict__ e_buf, const float* __restrict__ den,
                                const float* __restrict__ XH, float* __restrict__ agg) {
    const int FDIM = HEADS * CH;
    long i = (long)blockIdx.x * blockDim.x + threadIdx.x;
    if (i >= (long)ETOT * FDIM) return;
    int e = (int)(i / FDIM);
    int f = (int)(i % FDIM);          // f = h*CH + c
    int h = f / CH;
    int s, d; edge_nodes(e, src, dst, s, d);
    float alpha = e_buf[e * HEADS + h] / (den[d * HEADS + h] + 1e-16f);
    atomicAdd(&agg[(long)d * FDIM + f], XH[(long)s * FDIM + f] * alpha);
}

// ---------------- layer-1 epilogue: x1h = f16(relu(agg1 + b1)) ----------------
__global__ void relu_f16_kernel(const float* __restrict__ agg, const float* __restrict__ b,
                                _Float16* __restrict__ out) {
    int i = blockIdx.x * blockDim.x + threadIdx.x;
    if (i >= NNODES * F1) return;
    float v = agg[i] + b[i % F1];
    out[i] = (_Float16)(v > 0.f ? v : 0.f);
}

// ---------------- final: out = agg2 + b2  (H2==1 => mean over heads is identity) ----------------
__global__ void final_kernel(const float* __restrict__ agg, const float* __restrict__ b,
                             float* __restrict__ out) {
    int i = blockIdx.x * blockDim.x + threadIdx.x;
    if (i >= NNODES * CH) return;
    out[i] = agg[i] + b[i % CH];
}

static inline int nblk(long n) { return (int)((n + 255) / 256); }

extern "C" void kernel_launch(void* const* d_in, const int* in_sizes, int n_in,
                              void* d_out, int out_size, void* d_ws, size_t ws_size,
                              hipStream_t stream) {
    const float* feat = (const float*)d_in[0];
    const int*   eidx = (const int*)d_in[1];
    const int*   src  = eidx;
    const int*   dst  = eidx + NEDGES;
    const float* W1   = (const float*)d_in[2];
    const float* atS1 = (const float*)d_in[3];
    const float* atD1 = (const float*)d_in[4];
    const float* b1   = (const float*)d_in[5];
    const float* W2   = (const float*)d_in[6];
    const float* atS2 = (const float*)d_in[7];
    const float* atD2 = (const float*)d_in[8];
    const float* b2   = (const float*)d_in[9];
    float* out = (float*)d_out;

    char* ws = (char*)d_ws;
    // ---- layer-1 region (peak ~54.4 MB) ----
    float*    xh1   = (float*)(ws + 0);              // 20,480,000 B : N x 256
    float*    agg1  = (float*)(ws + 20480000);       // 20,480,000 B : N x 256
    float*    e1    = (float*)(ws + 40960000);       // 10,880,000 B : ETOT x 8
    _Float16* x1h   = (_Float16*)(ws + 40960000);    // overlays e1 (dead after agg pass)
    float*    asr1  = (float*)(ws + 51840000);       //   640,000 B : N x 8
    float*    adr1  = (float*)(ws + 52480000);
    float*    m1    = (float*)(ws + 53120000);
    float*    den1  = (float*)(ws + 53760000);
    // ---- layer-2 region overlays dead xh1 block ----
    float*    xh2   = (float*)(ws + 0);              // 2,560,000 B : N x 32
    float*    agg2  = (float*)(ws + 2560000);        // 2,560,000 B
    float*    e2    = (float*)(ws + 5120000);        // 1,360,000 B : ETOT x 1
    float*    asr2  = (float*)(ws + 6480000);        //    80,000 B
    float*    adr2  = (float*)(ws + 6560000);
    float*    m2    = (float*)(ws + 6640000);
    float*    den2  = (float*)(ws + 6720000);

    // ===== layer 1 =====
    fill_kernel<<<nblk(NNODES * NH1), 256, 0, stream>>>(m1,  -INFINITY, NNODES * NH1);
    fill_kernel<<<nblk(NNODES * NH1), 256, 0, stream>>>(den1, 0.0f,     NNODES * NH1);
    fill_kernel<<<nblk(NNODES * F1),  256, 0, stream>>>(agg1, 0.0f,     NNODES * F1);

    gemm1_wmma<<<(1250 * 16) / 8, 256, 0, stream>>>(feat, W1, xh1);

    alpha_kernel<NH1><<<nblk(NNODES * NH1), 256, 0, stream>>>(xh1, atS1, atD1, asr1, adr1);

    edge_max_kernel<NH1><<<nblk((long)ETOT * NH1), 256, 0, stream>>>(src, dst, asr1, adr1, e1, m1);
    edge_exp_kernel<NH1><<<nblk((long)ETOT * NH1), 256, 0, stream>>>(src, dst, e1, m1, den1);
    edge_agg_kernel<NH1><<<nblk((long)ETOT * F1), 256, 0, stream>>>(src, dst, e1, den1, xh1, agg1);

    relu_f16_kernel<<<nblk(NNODES * F1), 256, 0, stream>>>(agg1, b1, x1h);

    // ===== layer 2 =====
    gemm2_wmma<<<(1250 * 2 + 7) / 8, 256, 0, stream>>>(x1h, W2, xh2);

    fill_kernel<<<nblk(NNODES), 256, 0, stream>>>(m2,  -INFINITY, NNODES);
    fill_kernel<<<nblk(NNODES), 256, 0, stream>>>(den2, 0.0f,     NNODES);
    fill_kernel<<<nblk(NNODES * CH), 256, 0, stream>>>(agg2, 0.0f, NNODES * CH);

    alpha_kernel<1><<<nblk(NNODES), 256, 0, stream>>>(xh2, atS2, atD2, asr2, adr2);

    edge_max_kernel<1><<<nblk(ETOT), 256, 0, stream>>>(src, dst, asr2, adr2, e2, m2);
    edge_exp_kernel<1><<<nblk(ETOT), 256, 0, stream>>>(src, dst, e2, m2, den2);
    edge_agg_kernel<1><<<nblk((long)ETOT * CH), 256, 0, stream>>>(src, dst, e2, den2, xh2, agg2);

    final_kernel<<<nblk(NNODES * CH), 256, 0, stream>>>(agg2, b2, out);
}
